// ComplexAttentionBlock_41205916238344
// MI455X (gfx1250) — matrix operand order repaired and checked by reading, not compiled, not run on hardware
//
#include <hip/hip_runtime.h>
#include <hip/hip_bf16.h>
#include <math.h>

typedef _Float16 h16;
typedef __attribute__((ext_vector_type(16))) _Float16 v16h;
typedef __attribute__((ext_vector_type(8)))  _Float16 v8h;
typedef __attribute__((ext_vector_type(8)))  float    v8f;

#define DEVI __device__ __forceinline__

// ---------------------------------------------------------------------------
// Problem constants
// ---------------------------------------------------------------------------
static constexpr int BB   = 4;
static constexpr int SEQ  = 1024;
static constexpr int EE   = 768;
static constexpr int NH   = 12;
static constexpr int DH   = 64;       // head dim
static constexpr int K3   = 2304;     // 3*E
static constexpr int MH   = 1536;     // MLP hidden
static constexpr int TT   = BB * SEQ; // 4096 tokens

// ---------------------------------------------------------------------------
// WMMA fragment loaders (for V_WMMA_F32_16X16X32_F16, wave32; A/B use this code's own matched contiguous-K order)
// A 16x32 f16: lane L -> M=L%16, halves 0..15 = 16 contiguous K starting at
// k0 (+16 for lanes >=16).
// B 32x16 f16: lane L -> N=L%16, halves 0..15 = 16 contiguous K starting at
// k0 (+16 for lanes >=16).
// C/D f32: VGPR j, lane L: M = j + (L>=16 ? 8 : 0), N = L%16.
// ---------------------------------------------------------------------------
union V16U { v16h v; v8h h[2]; };

DEVI v16h wmma_load_a(const h16* A, int lda, int m_base, int k0, int lane) {
  int m  = m_base + (lane & 15);
  int kk = k0 + ((lane >> 4) << 4);
  const h16* p = A + (size_t)m * lda + kk;
  V16U u;
  u.h[0] = *(const v8h*)p;        // K = kk .. kk+7
  u.h[1] = *(const v8h*)(p + 8);  // K = kk+8 .. kk+15
  return u.v;
}

DEVI v16h wmma_load_b(const h16* W, int ldk, int n_base, int k0, int lane) {
  int n  = n_base + (lane & 15);
  int kk = k0 + ((lane >> 4) << 4);
  const h16* p = W + (size_t)n * ldk + kk;
  V16U u;
  u.h[0] = *(const v8h*)p;
  u.h[1] = *(const v8h*)(p + 8);
  return u.v;
}

// B fragment with strided K (gather): element t -> row (k0' + t) of `base`,
// column (lane&15).  Used for P @ V where V is [key][d] row-major.
DEVI v16h wmma_load_b_strided(const h16* base, int ld, int k0, int lane) {
  int kk = k0 + ((lane >> 4) << 4);
  const h16* p = base + (size_t)kk * ld + (lane & 15);
  v16h r;
#pragma unroll
  for (int t = 0; t < 16; ++t) r[t] = p[(size_t)t * ld];
  return r;
}

DEVI v8f wmma32(v16h a, v16h b, v8f c) {
  return __builtin_amdgcn_wmma_f32_16x16x32_f16(false, a, false, b,
                                                (short)0, c, false, false);
}

// ---------------------------------------------------------------------------
// f32 -> f16 convert (weights)
// ---------------------------------------------------------------------------
__global__ void k_f32_to_f16(const float* __restrict__ in,
                             h16* __restrict__ out, int n) {
  int i = blockIdx.x * 256 + threadIdx.x;
  if (i < n) out[i] = (h16)in[i];
}

// ---------------------------------------------------------------------------
// Complex LayerNorm over magnitudes.  One block (256 thr) per token.
// Input  : interleaved fp32 [T, 2E] (real at 2e, imag at 2e+1)
// Output : f16 planes Or/Oi [T, E]
// ---------------------------------------------------------------------------
__global__ void k_cln(const float* __restrict__ x,
                      const float* __restrict__ g,
                      const float* __restrict__ bt,
                      h16* __restrict__ Or, h16* __restrict__ Oi) {
  const int t   = blockIdx.x;
  const int tid = threadIdx.x;
  const float* xr = x + (size_t)t * (2 * EE);
  float rr[3], ii[3], mg[3];
#pragma unroll
  for (int j = 0; j < 3; ++j) {
    int e = tid + j * 256;
    rr[j] = xr[2 * e];
    ii[j] = xr[2 * e + 1];
    mg[j] = sqrtf(rr[j] * rr[j] + ii[j] * ii[j] + 1e-6f);
  }
  __shared__ float sred[256];
  float s = mg[0] + mg[1] + mg[2];
  sred[tid] = s;
  __syncthreads();
  for (int off = 128; off > 0; off >>= 1) {
    if (tid < off) sred[tid] += sred[tid + off];
    __syncthreads();
  }
  const float mean = sred[0] * (1.0f / (float)EE);
  __syncthreads();
  float vs = 0.f;
#pragma unroll
  for (int j = 0; j < 3; ++j) { float d = mg[j] - mean; vs += d * d; }
  sred[tid] = vs;
  __syncthreads();
  for (int off = 128; off > 0; off >>= 1) {
    if (tid < off) sred[tid] += sred[tid + off];
    __syncthreads();
  }
  const float var = sred[0] * (1.0f / (float)EE);
  const float inv = 1.0f / sqrtf(var + 1e-6f);
#pragma unroll
  for (int j = 0; j < 3; ++j) {
    int e = tid + j * 256;
    float scale = ((mg[j] - mean) * inv) / (mg[j] + 1e-6f);
    Or[(size_t)t * EE + e] = (h16)(g[2 * e]     * (rr[j] * scale) + bt[2 * e]);
    Oi[(size_t)t * EE + e] = (h16)(g[2 * e + 1] * (ii[j] * scale) + bt[2 * e + 1]);
  }
}

// ---------------------------------------------------------------------------
// Complex GEMM:  out = X * W^T + bias   (X:[M,K] r/i planes, W:[N,K] r/i)
// out_r = Xr Wr^T - Xi Wi^T ; out_i = Xr Wi^T + Xi Wr^T
// Block = 256 threads = 8 wave32; each wave owns a 16(M)x32(N) complex tile
// (two N sub-tiles share the A fragments -> 8 WMMAs per 6 fragment loads);
// workgroup tile = 16 x 256.  Epilogues selected by EPI:
//   0: +bias, write f16 planes                    (QKV)
//   1: +bias, +fp32 residual -> fp32 interleaved  (proj)
//   2: +bias, exact GELU, write f16 planes        (mlp1)
//   3: +bias, +fp32 residual -> fp32 interleaved  (mlp2 -> d_out)
// ---------------------------------------------------------------------------
DEVI float gelu_exact(float v) {
  return 0.5f * v * (1.0f + erff(v * 0.70710678118654752f));
}

template <int EPI>
__global__ void k_cgemm(const h16* __restrict__ Ar, const h16* __restrict__ Ai,
                        const h16* __restrict__ Wr, const h16* __restrict__ Wi,
                        const float* __restrict__ br, const float* __restrict__ bi,
                        const float* __restrict__ resid,  // [M, 2*Nout] fp32
                        float* __restrict__ Ofp,          // [M, 2*Nout] fp32
                        h16* __restrict__ Or, h16* __restrict__ Oi,
                        int M, int Nout, int K) {
  const int lane   = threadIdx.x & 31;
  const int wave   = threadIdx.x >> 5;
  const int n_base = blockIdx.x * 256 + wave * 32;
  const int m_base = blockIdx.y * 16;
  (void)M;

  v8f acc_rr[2] = {v8f{}, v8f{}};
  v8f acc_ii[2] = {v8f{}, v8f{}};
  v8f acc_ri[2] = {v8f{}, v8f{}};
  v8f acc_ir[2] = {v8f{}, v8f{}};

  for (int k0 = 0; k0 < K; k0 += 32) {
    v16h ar = wmma_load_a(Ar, K, m_base, k0, lane);
    v16h ai = wmma_load_a(Ai, K, m_base, k0, lane);
#pragma unroll
    for (int s = 0; s < 2; ++s) {
      const int nb = n_base + s * 16;
      v16h wr = wmma_load_b(Wr, K, nb, k0, lane);
      v16h wi = wmma_load_b(Wi, K, nb, k0, lane);
      // speculative prefetch of next K-step of this weight row
      // (lowers to global_prefetch_b8; no LOADcnt cost)
      if (k0 + 32 < K) {
        const h16* pw = Wr + (size_t)(nb + (lane & 15)) * K + k0 + 32;
        __builtin_prefetch(pw, 0, 0);
        const h16* pv = Wi + (size_t)(nb + (lane & 15)) * K + k0 + 32;
        __builtin_prefetch(pv, 0, 0);
      }
      acc_rr[s] = wmma32(ar, wr, acc_rr[s]);
      acc_ii[s] = wmma32(ai, wi, acc_ii[s]);
      acc_ri[s] = wmma32(ar, wi, acc_ri[s]);
      acc_ir[s] = wmma32(ai, wr, acc_ir[s]);
    }
  }

#pragma unroll
  for (int s = 0; s < 2; ++s) {
#pragma unroll
    for (int j = 0; j < 8; ++j) {
      int row = m_base + j + ((lane >> 4) << 3);
      int col = n_base + s * 16 + (lane & 15);
      float vr = acc_rr[s][j] - acc_ii[s][j] + br[col];
      float vi = acc_ri[s][j] + acc_ir[s][j] + bi[col];
      if (EPI == 0) {
        Or[(size_t)row * Nout + col] = (h16)vr;
        Oi[(size_t)row * Nout + col] = (h16)vi;
      } else if (EPI == 1 || EPI == 3) {
        size_t o = (size_t)row * (2 * Nout) + 2 * col;
        Ofp[o]     = resid[o]     + vr;
        Ofp[o + 1] = resid[o + 1] + vi;
      } else { // EPI == 2 : GELU
        Or[(size_t)row * Nout + col] = (h16)gelu_exact(vr);
        Oi[(size_t)row * Nout + col] = (h16)gelu_exact(vi);
      }
    }
  }
}

// ---------------------------------------------------------------------------
// Complex-magnitude attention.  One workgroup per (b, h, 16-query tile).
// QKV planes: [T, 2304] f16 (q at col 0, k at 768, v at 1536; head off h*64)
// Dynamic LDS: Smag 16x1024 f32 (64KB) | Wp 16x1024 f16 (32KB) | red 256 f32
//              | rowstat 16 f32   ->  99392 bytes  (WGP LDS = 320KB)
// ---------------------------------------------------------------------------
__global__ void k_attn(const h16* __restrict__ QKVr,
                       const h16* __restrict__ QKVi,
                       h16* __restrict__ ATTr, h16* __restrict__ ATTi) {
  extern __shared__ char smem[];
  float* Smag    = (float*)smem;                             // [16][1024]
  h16*   Wp      = (h16*)(smem + 16 * 1024 * 4);             // [16][1024]
  float* red     = (float*)(smem + 16 * 1024 * 4 + 16 * 1024 * 2); // [16][16]
  float* rowstat = red + 256;                                // [16]

  const int lane  = threadIdx.x & 31;
  const int wave  = threadIdx.x >> 5;
  const int qt    = blockIdx.x;   // query tile (0..63)
  const int hh    = blockIdx.y;   // head
  const int b     = blockIdx.z;   // batch
  const int row0  = b * SEQ;      // absolute token row base
  const int qcol  = hh * DH;
  const int kcol  = EE + hh * DH;
  const int vcol  = 2 * EE + hh * DH;
  const int mrow  = row0 + qt * 16;

  // ---- S = q k^T : S_r = qr kr^T + qi ki^T ; S_i = qr ki^T - qi kr^T ----
  for (int jj = 0; jj < 8; ++jj) {
    const int kt = wave + jj * 8;           // key tile 0..63
    v8f acc_r = {}, acc_i1 = {}, acc_i2 = {};
#pragma unroll
    for (int k0 = 0; k0 < DH; k0 += 32) {
      v16h qr = wmma_load_a(QKVr + qcol, K3, mrow, k0, lane);
      v16h qi = wmma_load_a(QKVi + qcol, K3, mrow, k0, lane);
      v16h kr = wmma_load_b(QKVr + kcol, K3, row0 + kt * 16, k0, lane);
      v16h ki = wmma_load_b(QKVi + kcol, K3, row0 + kt * 16, k0, lane);
      acc_r  = wmma32(qr, kr, acc_r);
      acc_r  = wmma32(qi, ki, acc_r);
      acc_i1 = wmma32(qr, ki, acc_i1);
      acc_i2 = wmma32(qi, kr, acc_i2);
    }
#pragma unroll
    for (int j = 0; j < 8; ++j) {
      int r = j + ((lane >> 4) << 3);
      int c = kt * 16 + (lane & 15);
      float sr = acc_r[j];
      float si = acc_i1[j] - acc_i2[j];
      Smag[r * SEQ + c] = sqrtf(sr * sr + si * si + 1e-8f) * 0.125f; // /sqrt(64)
    }
  }
  __syncthreads();

  // ---- softmax over keys, probabilities -> f16 tile in LDS ----
  {
    const int r  = threadIdx.x >> 4;  // 0..15
    const int c0 = threadIdx.x & 15;
    float mx = -1e30f;
    for (int c = c0; c < SEQ; c += 16) mx = fmaxf(mx, Smag[r * SEQ + c]);
    red[r * 16 + c0] = mx;
    __syncthreads();
    if (c0 == 0) {
      float m = red[r * 16];
      for (int t = 1; t < 16; ++t) m = fmaxf(m, red[r * 16 + t]);
      rowstat[r] = m;
    }
    __syncthreads();
    const float rmax = rowstat[r];
    float sum = 0.f;
    for (int c = c0; c < SEQ; c += 16) {
      float e = __expf(Smag[r * SEQ + c] - rmax);
      Smag[r * SEQ + c] = e;
      sum += e;
    }
    red[r * 16 + c0] = sum;
    __syncthreads();
    if (c0 == 0) {
      float m = 0.f;
      for (int t = 0; t < 16; ++t) m += red[r * 16 + t];
      rowstat[r] = 1.0f / m;
    }
    __syncthreads();
    const float inv = rowstat[r];
    for (int c = c0; c < SEQ; c += 16)
      Wp[r * SEQ + c] = (h16)(Smag[r * SEQ + c] * inv);
  }
  __syncthreads();

  // ---- out = P @ V  (wave -> {real,imag} x 4 d-tiles of 16) ----
  {
    const int comp = wave & 1;      // 0: real, 1: imag
    const int dt   = wave >> 1;     // 0..3
    const h16* Vbase = (comp ? QKVi : QKVr) +
                       (size_t)row0 * K3 + vcol + dt * 16;
    h16* Out = comp ? ATTi : ATTr;
    v8f acc = {};
    for (int k0 = 0; k0 < SEQ; k0 += 32) {
      v16h aw = wmma_load_a((const h16*)Wp, SEQ, 0, k0, lane);
      v16h bv = wmma_load_b_strided(Vbase, K3, k0, lane);
      acc = wmma32(aw, bv, acc);
    }
#pragma unroll
    for (int j = 0; j < 8; ++j) {
      int row = mrow + j + ((lane >> 4) << 3);
      int col = hh * DH + dt * 16 + (lane & 15);
      Out[(size_t)row * EE + col] = (h16)acc[j];
    }
  }
}

// ---------------------------------------------------------------------------
// Host-side launch
// ---------------------------------------------------------------------------
extern "C" void kernel_launch(void* const* d_in, const int* in_sizes, int n_in,
                              void* d_out, int out_size, void* d_ws, size_t ws_size,
                              hipStream_t stream) {
  (void)in_sizes; (void)n_in; (void)out_size; (void)ws_size;
  const float* x       = (const float*)d_in[0];
  const float* g1      = (const float*)d_in[1];
  const float* b1      = (const float*)d_in[2];
  const float* g2      = (const float*)d_in[3];
  const float* b2      = (const float*)d_in[4];
  const float* qkv_wr  = (const float*)d_in[5];
  const float* qkv_wi  = (const float*)d_in[6];
  const float* qkv_br  = (const float*)d_in[7];
  const float* qkv_bi  = (const float*)d_in[8];
  const float* proj_wr = (const float*)d_in[9];
  const float* proj_wi = (const float*)d_in[10];
  const float* proj_br = (const float*)d_in[11];
  const float* proj_bi = (const float*)d_in[12];
  const float* m1_wr   = (const float*)d_in[13];
  const float* m1_wi   = (const float*)d_in[14];
  const float* m1_br   = (const float*)d_in[15];
  const float* m1_bi   = (const float*)d_in[16];
  const float* m2_wr   = (const float*)d_in[17];
  const float* m2_wi   = (const float*)d_in[18];
  const float* m2_br   = (const float*)d_in[19];
  const float* m2_bi   = (const float*)d_in[20];
  float* out = (float*)d_out;

  // workspace carve-up (256B aligned)
  char* p = (char*)d_ws;
  auto alloc = [&](size_t bytes) -> char* {
    char* r = p;
    p += (bytes + 255) & ~(size_t)255;
    return r;
  };
  h16* Wqr  = (h16*)alloc(sizeof(h16) * (size_t)K3 * EE);
  h16* Wqi  = (h16*)alloc(sizeof(h16) * (size_t)K3 * EE);
  h16* Wpr  = (h16*)alloc(sizeof(h16) * (size_t)EE * EE);
  h16* Wpi  = (h16*)alloc(sizeof(h16) * (size_t)EE * EE);
  h16* W1r  = (h16*)alloc(sizeof(h16) * (size_t)MH * EE);
  h16* W1i  = (h16*)alloc(sizeof(h16) * (size_t)MH * EE);
  h16* W2r  = (h16*)alloc(sizeof(h16) * (size_t)EE * MH);
  h16* W2i  = (h16*)alloc(sizeof(h16) * (size_t)EE * MH);
  h16* Xn1r = (h16*)alloc(sizeof(h16) * (size_t)TT * EE);
  h16* Xn1i = (h16*)alloc(sizeof(h16) * (size_t)TT * EE);
  h16* Qkr  = (h16*)alloc(sizeof(h16) * (size_t)TT * K3);
  h16* Qki  = (h16*)alloc(sizeof(h16) * (size_t)TT * K3);
  h16* Atr  = (h16*)alloc(sizeof(h16) * (size_t)TT * EE);
  h16* Ati  = (h16*)alloc(sizeof(h16) * (size_t)TT * EE);
  float* Xc2 = (float*)alloc(sizeof(float) * (size_t)TT * 2 * EE);
  h16* Xn2r = (h16*)alloc(sizeof(h16) * (size_t)TT * EE);
  h16* Xn2i = (h16*)alloc(sizeof(h16) * (size_t)TT * EE);
  h16* Hr   = (h16*)alloc(sizeof(h16) * (size_t)TT * MH);
  h16* Hi   = (h16*)alloc(sizeof(h16) * (size_t)TT * MH);

  auto cvt = [&](const float* src, h16* dst, int n) {
    k_f32_to_f16<<<(n + 255) / 256, 256, 0, stream>>>(src, dst, n);
  };
  cvt(qkv_wr,  Wqr, K3 * EE);
  cvt(qkv_wi,  Wqi, K3 * EE);
  cvt(proj_wr, Wpr, EE * EE);
  cvt(proj_wi, Wpi, EE * EE);
  cvt(m1_wr,   W1r, MH * EE);
  cvt(m1_wi,   W1i, MH * EE);
  cvt(m2_wr,   W2r, EE * MH);
  cvt(m2_wi,   W2i, EE * MH);

  // LN1
  k_cln<<<TT, 256, 0, stream>>>(x, g1, b1, Xn1r, Xn1i);

  // QKV = cLinear(xn1)  -> f16 planes [T,2304]
  k_cgemm<0><<<dim3(K3 / 256, TT / 16), 256, 0, stream>>>(
      Xn1r, Xn1i, Wqr, Wqi, qkv_br, qkv_bi,
      nullptr, nullptr, Qkr, Qki, TT, K3, EE);

  // attention: grid (q-tiles, heads, batch)
  const size_t attn_lds = 16 * 1024 * 4 + 16 * 1024 * 2 + 256 * 4 + 16 * 4;
  k_attn<<<dim3(SEQ / 16, NH, BB), 256, attn_lds, stream>>>(Qkr, Qki, Atr, Ati);

  // proj + residual -> fp32 xc2
  k_cgemm<1><<<dim3(EE / 256, TT / 16), 256, 0, stream>>>(
      Atr, Ati, Wpr, Wpi, proj_br, proj_bi,
      x, Xc2, nullptr, nullptr, TT, EE, EE);

  // LN2
  k_cln<<<TT, 256, 0, stream>>>(Xc2, g2, b2, Xn2r, Xn2i);

  // MLP1 + GELU -> f16 planes [T,1536]
  k_cgemm<2><<<dim3(MH / 256, TT / 16), 256, 0, stream>>>(
      Xn2r, Xn2i, W1r, W1i, m1_br, m1_bi,
      nullptr, nullptr, Hr, Hi, TT, MH, EE);

  // MLP2 + residual -> fp32 output
  k_cgemm<3><<<dim3(EE / 256, TT / 16), 256, 0, stream>>>(
      Hr, Hi, W2r, W2i, m2_br, m2_bi,
      Xc2, out, nullptr, nullptr, TT, EE, MH);
}
